// GGNNLayer_7172595384548
// MI455X (gfx1250) — compile-verified, matrix operand order checked by reading, not run on hardware
//
#include <hip/hip_runtime.h>

#define DD 128

typedef __attribute__((ext_vector_type(16))) __bf16 v16bf;
typedef __attribute__((ext_vector_type(8)))  float  v8f;

// ---------- bf16 helpers (hi/lo split for fp32-accurate WMMA) ----------
static __device__ __forceinline__ unsigned short f2bf(float f) {
  unsigned int u = __float_as_uint(f);
  unsigned int r = (u + 0x7FFFu + ((u >> 16) & 1u)) >> 16;  // round-nearest-even
  return (unsigned short)r;
}
static __device__ __forceinline__ float bf2f(unsigned short h) {
  return __uint_as_float(((unsigned int)h) << 16);
}

union FragU { v16bf v; uint4 q[2]; };

static __device__ __forceinline__ v16bf frag_from(const unsigned short* p) {
  FragU u;
  u.q[0] = *(const uint4*)(p);
  u.q[1] = *(const uint4*)(p + 16);
  return u.v;
}
// A fragment from a row-major 16x128 bf16 tile: lane l -> row l&15,
// K elements {kb..kb+7, 16+kb..16+kb+7}, kb = (l>>4)*8  (ISA 7.12.2 layout)
static __device__ __forceinline__ v16bf load_a(const unsigned short* tile, int lane, int kc) {
  int row = lane & 15;
  int kb  = (lane >> 4) << 3;
  return frag_from(tile + row * DD + kc * 32 + kb);
}
// B fragment (output cols nBase..nBase+15) from row-major [384 x 128] weight,
// row index = output column -> contiguous in K, mirrored lane layout.
static __device__ __forceinline__ v16bf load_b(const unsigned short* wmat, int lane, int kc, int nBase) {
  int col = nBase + (lane & 15);
  int kb  = (lane >> 4) << 3;
  return frag_from(wmat + col * DD + kc * 32 + kb);
}

#define WMMA_BF16(acc, A, B) \
  acc = __builtin_amdgcn_wmma_f32_16x16x32_bf16(false, (A), false, (B), (short)0, (acc), false, false)

// ---------- kernel 1: zero the atomic accumulators ----------
__global__ void k_zero_f32(float* p, long long n) {
  long long stride = (long long)gridDim.x * blockDim.x;
  for (long long i = (long long)blockIdx.x * blockDim.x + threadIdx.x; i < n; i += stride)
    p[i] = 0.0f;
}

// ---------- kernel 2: build combined weights Wc1 = w_ih[:, :128]@W1,
// Wc2 = w_ih[:,128:]@W2, and split Wc1/Wc2/w_hh into bf16 hi/lo ----------
__global__ void k_prep_weights(const float* __restrict__ W1, const float* __restrict__ W2,
                               const float* __restrict__ w_ih, const float* __restrict__ w_hh,
                               unsigned short* wc1h, unsigned short* wc1l,
                               unsigned short* wc2h, unsigned short* wc2l,
                               unsigned short* whhh, unsigned short* whhl) {
  int t = blockIdx.x * blockDim.x + threadIdx.x;
  const int PER = 384 * DD;
  if (t >= 3 * PER) return;
  int m = t / PER;
  int r = t - m * PER;
  int o = r / DD;      // output row (0..383)
  int k = r - o * DD;  // input col (0..127)
  float acc = 0.0f;
  unsigned short *ph, *pl;
  if (m == 0) {
    for (int j = 0; j < DD; ++j) acc += w_ih[o * 256 + j] * W1[j * DD + k];
    ph = wc1h; pl = wc1l;
  } else if (m == 1) {
    for (int j = 0; j < DD; ++j) acc += w_ih[o * 256 + 128 + j] * W2[j * DD + k];
    ph = wc2h; pl = wc2l;
  } else {
    acc = w_hh[o * DD + k];
    ph = whhh; pl = whhl;
  }
  unsigned short h = f2bf(acc);
  ph[r] = h;
  pl[r] = f2bf(acc - bf2f(h));
}

// ---------- kernel 3: edge scatter (weighted message sums, both directions) ----------
__global__ void k_scatter(const float* __restrict__ feat, const float* __restrict__ w,
                          const int* __restrict__ src, const int* __restrict__ dst,
                          float* msum1, float* wsum1, float* msum2, float* wsum2, int E) {
  int idx = blockIdx.x * blockDim.x + threadIdx.x;
  if (idx >= E * 32) return;
  int e = idx >> 5;
  int c = (idx & 31) << 2;  // 4-float chunk
  float ww = w[e];
  int s = src[e], d = dst[e];
  float4 f1 = *(const float4*)(feat + (size_t)s * DD + c);
  atomicAdd(&msum1[(size_t)d * DD + c + 0], f1.x * ww);
  atomicAdd(&msum1[(size_t)d * DD + c + 1], f1.y * ww);
  atomicAdd(&msum1[(size_t)d * DD + c + 2], f1.z * ww);
  atomicAdd(&msum1[(size_t)d * DD + c + 3], f1.w * ww);
  float4 f2 = *(const float4*)(feat + (size_t)d * DD + c);
  atomicAdd(&msum2[(size_t)s * DD + c + 0], f2.x * ww);
  atomicAdd(&msum2[(size_t)s * DD + c + 1], f2.y * ww);
  atomicAdd(&msum2[(size_t)s * DD + c + 2], f2.z * ww);
  atomicAdd(&msum2[(size_t)s * DD + c + 3], f2.w * ww);
  if ((idx & 31) == 0) {
    atomicAdd(&wsum1[d], ww);
    atomicAdd(&wsum2[s], ww);
  }
}

// ---------- kernel 4: weighted mean + bf16 hi/lo split of neigh1/neigh2/feat ----------
__global__ void k_finalize(const float* __restrict__ feat,
                           const float* __restrict__ msum1, const float* __restrict__ msum2,
                           const float* __restrict__ wsum1, const float* __restrict__ wsum2,
                           unsigned short* n1h, unsigned short* n1l,
                           unsigned short* n2h, unsigned short* n2l,
                           unsigned short* fh, unsigned short* fl,
                           int N, long long total) {
  long long t = (long long)blockIdx.x * blockDim.x + threadIdx.x;
  if (t >= total) return;
  int n = (int)(t >> 7);
  float w1 = wsum1[n];
  float v1 = (w1 > 0.0f) ? msum1[t] / w1 : 0.0f;
  unsigned short h1 = f2bf(v1);
  n1h[t] = h1; n1l[t] = f2bf(v1 - bf2f(h1));
  float w2 = wsum2[n];
  float v2 = (w2 > 0.0f) ? msum2[t] / w2 : 0.0f;
  unsigned short h2 = f2bf(v2);
  n2h[t] = h2; n2l[t] = f2bf(v2 - bf2f(h2));
  float f = (n < N) ? feat[t] : 0.0f;
  unsigned short hf = f2bf(f);
  fh[t] = hf; fl[t] = f2bf(f - bf2f(hf));
}

// ---------- kernel 5: fused WMMA GEMMs + GRU gates ----------
// One block = one 16-node tile; A tiles staged in LDS; wave w owns output
// columns [16w, 16w+16) and keeps i_r,i_z,i_n,h_r,h_z,h_n accumulators live.
__global__ __launch_bounds__(256) void k_gemm_gates(
    const unsigned short* __restrict__ n1h, const unsigned short* __restrict__ n1l,
    const unsigned short* __restrict__ n2h, const unsigned short* __restrict__ n2l,
    const unsigned short* __restrict__ fh,  const unsigned short* __restrict__ fl,
    const unsigned short* __restrict__ wc1h, const unsigned short* __restrict__ wc1l,
    const unsigned short* __restrict__ wc2h, const unsigned short* __restrict__ wc2l,
    const unsigned short* __restrict__ whhh, const unsigned short* __restrict__ whhl,
    const float* __restrict__ b_ih, const float* __restrict__ b_hh,
    float* __restrict__ out, int N) {
  __shared__ __align__(16) unsigned short sA[6][16 * DD];

  const int mTile = blockIdx.x;
  const size_t base = (size_t)mTile * 16 * DD;
  const unsigned int* srcs[6] = {
      (const unsigned int*)(n1h + base), (const unsigned int*)(n1l + base),
      (const unsigned int*)(n2h + base), (const unsigned int*)(n2l + base),
      (const unsigned int*)(fh + base),  (const unsigned int*)(fl + base)};
  for (int i = threadIdx.x; i < 6 * 1024; i += 256) {  // 16*128 bf16 = 1024 dwords
    int t = i >> 10, j = i & 1023;
    ((unsigned int*)sA[t])[j] = srcs[t][j];
  }
  __syncthreads();

  const int lane  = threadIdx.x & 31;
  const int cBase = (threadIdx.x >> 5) << 4;  // wave's 16-column output slice

  v8f ir = {}, iz = {}, inn = {}, hr = {}, hz = {}, hn = {};
  for (int kc = 0; kc < 4; ++kc) {
    v16bf a1h = load_a(sA[0], lane, kc), a1l = load_a(sA[1], lane, kc);
    v16bf a2h = load_a(sA[2], lane, kc), a2l = load_a(sA[3], lane, kc);
    v16bf afh = load_a(sA[4], lane, kc), afl = load_a(sA[5], lane, kc);
    // i_r / i_z / i_n : neigh1 x Wc1 + neigh2 x Wc2 (hi/lo split => 3 WMMAs each)
    {
      v16bf bh = load_b(wc1h, lane, kc, cBase),       bl = load_b(wc1l, lane, kc, cBase);
      WMMA_BF16(ir, a1h, bh); WMMA_BF16(ir, a1h, bl); WMMA_BF16(ir, a1l, bh);
      bh = load_b(wc2h, lane, kc, cBase);             bl = load_b(wc2l, lane, kc, cBase);
      WMMA_BF16(ir, a2h, bh); WMMA_BF16(ir, a2h, bl); WMMA_BF16(ir, a2l, bh);
    }
    {
      v16bf bh = load_b(wc1h, lane, kc, 128 + cBase), bl = load_b(wc1l, lane, kc, 128 + cBase);
      WMMA_BF16(iz, a1h, bh); WMMA_BF16(iz, a1h, bl); WMMA_BF16(iz, a1l, bh);
      bh = load_b(wc2h, lane, kc, 128 + cBase);       bl = load_b(wc2l, lane, kc, 128 + cBase);
      WMMA_BF16(iz, a2h, bh); WMMA_BF16(iz, a2h, bl); WMMA_BF16(iz, a2l, bh);
    }
    {
      v16bf bh = load_b(wc1h, lane, kc, 256 + cBase), bl = load_b(wc1l, lane, kc, 256 + cBase);
      WMMA_BF16(inn, a1h, bh); WMMA_BF16(inn, a1h, bl); WMMA_BF16(inn, a1l, bh);
      bh = load_b(wc2h, lane, kc, 256 + cBase);       bl = load_b(wc2l, lane, kc, 256 + cBase);
      WMMA_BF16(inn, a2h, bh); WMMA_BF16(inn, a2h, bl); WMMA_BF16(inn, a2l, bh);
    }
    // h_r / h_z / h_n : feat x w_hh
    {
      v16bf bh = load_b(whhh, lane, kc, cBase),       bl = load_b(whhl, lane, kc, cBase);
      WMMA_BF16(hr, afh, bh); WMMA_BF16(hr, afh, bl); WMMA_BF16(hr, afl, bh);
      bh = load_b(whhh, lane, kc, 128 + cBase);       bl = load_b(whhl, lane, kc, 128 + cBase);
      WMMA_BF16(hz, afh, bh); WMMA_BF16(hz, afh, bl); WMMA_BF16(hz, afl, bh);
      bh = load_b(whhh, lane, kc, 256 + cBase);       bl = load_b(whhl, lane, kc, 256 + cBase);
      WMMA_BF16(hn, afh, bh); WMMA_BF16(hn, afh, bl); WMMA_BF16(hn, afl, bh);
    }
  }

  // GRU gates on in-register accumulators.
  // C/D layout: lane l, vgpr r -> row = r + (l>=16 ? 8 : 0), col = l&15.
  const int col  = lane & 15;
  const int rofs = (lane >> 4) << 3;
  const int gcol = cBase + col;
  const float bir = b_ih[gcol], biz = b_ih[128 + gcol], bin = b_ih[256 + gcol];
  const float bhr = b_hh[gcol], bhz = b_hh[128 + gcol], bhn = b_hh[256 + gcol];
#pragma unroll
  for (int r = 0; r < 8; ++r) {
    int row  = r + rofs;
    int node = mTile * 16 + row;
    float rg  = 1.0f / (1.0f + __expf(-(ir[r] + bir + hr[r] + bhr)));
    float zg  = 1.0f / (1.0f + __expf(-(iz[r] + biz + hz[r] + bhz)));
    float nnv = tanhf(inn[r] + bin + rg * (hn[r] + bhn));
    float h   = bf2f(sA[4][row * DD + gcol]) + bf2f(sA[5][row * DD + gcol]);
    if (node < N) out[(size_t)node * DD + gcol] = nnv + zg * (h - nnv);
  }
}

// ---------- launch ----------
extern "C" void kernel_launch(void* const* d_in, const int* in_sizes, int n_in,
                              void* d_out, int out_size, void* d_ws, size_t ws_size,
                              hipStream_t stream) {
  const float* feat = (const float*)d_in[0];
  const float* w    = (const float*)d_in[1];
  const int*   src  = (const int*)d_in[2];
  const int*   dst  = (const int*)d_in[3];
  const float* W1   = (const float*)d_in[4];
  const float* W2   = (const float*)d_in[5];
  const float* w_ih = (const float*)d_in[6];
  const float* w_hh = (const float*)d_in[7];
  const float* b_ih = (const float*)d_in[8];
  const float* b_hh = (const float*)d_in[9];

  const int N    = in_sizes[0] / DD;
  const int E    = in_sizes[1];
  const int Npad = (N + 15) & ~15;
  const size_t NF = (size_t)Npad * DD;
  const size_t WSZ = (size_t)384 * DD;

  // workspace layout (all 16B-aligned; ~258 MB total)
  char* p = (char*)d_ws;
  float* msum1 = (float*)p;          p += NF * 4;
  float* msum2 = (float*)p;          p += NF * 4;
  float* wsum1 = (float*)p;          p += (size_t)Npad * 4;
  float* wsum2 = (float*)p;          p += (size_t)Npad * 4;
  unsigned short* n1h = (unsigned short*)p;  p += NF * 2;
  unsigned short* n1l = (unsigned short*)p;  p += NF * 2;
  unsigned short* n2h = (unsigned short*)p;  p += NF * 2;
  unsigned short* n2l = (unsigned short*)p;  p += NF * 2;
  unsigned short* fh  = (unsigned short*)p;  p += NF * 2;
  unsigned short* fl  = (unsigned short*)p;  p += NF * 2;
  unsigned short* wc1h = (unsigned short*)p; p += WSZ * 2;
  unsigned short* wc1l = (unsigned short*)p; p += WSZ * 2;
  unsigned short* wc2h = (unsigned short*)p; p += WSZ * 2;
  unsigned short* wc2l = (unsigned short*)p; p += WSZ * 2;
  unsigned short* whhh = (unsigned short*)p; p += WSZ * 2;
  unsigned short* whhl = (unsigned short*)p; p += WSZ * 2;

  // 1) zero atomic accumulators (msum1,msum2,wsum1,wsum2 contiguous)
  long long zn = (long long)NF * 2 + (long long)Npad * 2;
  k_zero_f32<<<2048, 256, 0, stream>>>(msum1, zn);

  // 2) combined weights + bf16 hi/lo split
  k_prep_weights<<<(int)((3 * WSZ + 255) / 256), 256, 0, stream>>>(
      W1, W2, w_ih, w_hh, wc1h, wc1l, wc2h, wc2l, whhh, whhl);

  // 3) edge scatter (both graph directions)
  k_scatter<<<(E * 32 + 255) / 256, 256, 0, stream>>>(
      feat, w, src, dst, msum1, wsum1, msum2, wsum2, E);

  // 4) weighted mean + bf16 split of A matrices
  k_finalize<<<(int)((NF + 255) / 256), 256, 0, stream>>>(
      feat, msum1, msum2, wsum1, wsum2, n1h, n1l, n2h, n2l, fh, fl, N, (long long)NF);

  // 5) fused WMMA GEMMs + GRU gates
  k_gemm_gates<<<Npad / 16, 256, 0, stream>>>(
      n1h, n1l, n2h, n2l, fh, fl, wc1h, wc1l, wc2h, wc2l, whhh, whhl,
      b_ih, b_hh, (float*)d_out, N);
}
